// QuestionEmbedding_20091857011386
// MI455X (gfx1250) — matrix operand order repaired and checked by reading, not compiled, not run on hardware
//
#include <hip/hip_runtime.h>
#include <hip/hip_bf16.h>
#include <cstdint>
#include <cstddef>

// Problem constants (from reference): B=512, W=40, V=12000, E=512, H=512
#define B_  512
#define W_  40
#define V_  12000
#define E_  512
#define H_  512
#define G4  2048   // 4*H

typedef __attribute__((ext_vector_type(16))) __bf16 v16bf;
typedef __attribute__((ext_vector_type(8)))  float  v8f;

union FragU { unsigned int u[8]; v16bf v; };

__device__ __forceinline__ float sigf(float x) { return 1.0f / (1.0f + __expf(-x)); }

// ---- WMMA fragment loaders from LDS (row-major [row][LDT] bf16, LDT=40) ----
// A 16x32 bf16 layout (ISA 7.12.2): lane L: row=L&15, khalf=(L>>4)*8;
//   VGPR v<4 -> K = khalf + 2v ; v>=4 -> K = 16 + khalf + 2(v-4)
__device__ __forceinline__ v16bf lds_frag_a(const __bf16* base) {
  const int lane = threadIdx.x & 31;
  const __bf16* p = base + (lane & 15) * 40 + ((lane >> 4) << 3);
  FragU f;
#pragma unroll
  for (int v = 0; v < 8; ++v) {
    int k = ((v & 4) << 2) + ((v & 3) << 1);
    f.u[v] = *reinterpret_cast<const unsigned int*>(p + k);
  }
  return f.v;
}
// B 32x16 bf16 layout: lane L: col=L&15, khalf=(L>>4)*8; same K mapping as A
__device__ __forceinline__ v16bf lds_frag_b(const __bf16* base) {
  const int lane = threadIdx.x & 31;
  const __bf16* p = base + (lane & 15) * 40 + ((lane >> 4) << 3);
  FragU f;
#pragma unroll
  for (int v = 0; v < 8; ++v)
    f.u[v] = *reinterpret_cast<const unsigned int*>(p + ((v & 4) << 2) + ((v & 3) << 1));
  return f.v;
}

// ---------------- prep kernels ----------------
__global__ void f32_to_bf16_kernel(const float* __restrict__ in, __bf16* __restrict__ out, int n) {
  int i = blockIdx.x * blockDim.x + threadIdx.x;
  if (i < n) out[i] = (__bf16)in[i];
}

__global__ void bias_combine_kernel(const float* __restrict__ b, float* __restrict__ o) {
  int i = blockIdx.x * blockDim.x + threadIdx.x;
  if (i < 2 * G4) {
    int d = i / G4, n = i % G4;
    o[i] = b[(size_t)d * 2 * G4 + n] + b[(size_t)d * 2 * G4 + G4 + n];
  }
}

__global__ void zero_u32_kernel(unsigned int* __restrict__ p, int n) {
  int i = blockIdx.x * blockDim.x + threadIdx.x;
  if (i < n) p[i] = 0u;
}

// emb[b,t,e] = tanh(table[idx][e]); table row0 = zeros, else lookup_W[e, idx-1]
__global__ __launch_bounds__(256) void embed_kernel(const int* __restrict__ qv,
                                                    const float* __restrict__ lut,
                                                    __bf16* __restrict__ emb) {
  const int bw = blockIdx.x;           // 0 .. B*W-1
  const int idx = qv[bw];
#pragma unroll
  for (int j = 0; j < 2; ++j) {
    int e = threadIdx.x + j * 256;
    float v = 0.0f;
    if (idx > 0) v = tanhf(lut[(size_t)e * V_ + (idx - 1)]);
    emb[(size_t)bw * E_ + e] = (__bf16)v;
  }
}

// ---------------- fused LSTM step: G = x_t@wih^T + h@whh^T + b; gates; h,c update ----------------
// grid: (8 nh-tiles of 64 h-cols, 4 m-tiles of 128 rows, 2 dirs); block 256 = 8 waves
__global__ __launch_bounds__(256) void lstm_step_kernel(
    const __bf16* __restrict__ x, int x_tstride, int x_ld, int K1,
    const __bf16* __restrict__ wih, const __bf16* __restrict__ whh,
    const float* __restrict__ bias,
    const __bf16* __restrict__ h_in, __bf16* __restrict__ h_out,
    float* __restrict__ c_buf,
    __bf16* __restrict__ out_bf,   // layer-0: (B,W,2H) bf16; else null
    float* __restrict__ out_f,     // layer-1: (B,2H) f32 final; else null
    const int* __restrict__ qlen, int step)
{
  constexpr int LDT = 40;                     // 32 + 8 pad (bank-conflict break)
  __shared__ __bf16 lds_a[128 * LDT];         // activation tile 128x32
  __shared__ __bf16 lds_b[256 * LDT];         // weight tile: 4 gates x 64 rows x 32

  const int dir = blockIdx.z;
  const int t   = dir ? (W_ - 1 - step) : step;
  const int nh_base = blockIdx.x * 64;
  const int bm      = blockIdx.y * 128;
  const int tid  = threadIdx.x;
  const int lane = tid & 31;
  const int wid  = tid >> 5;
  const int mw   = wid >> 2;                  // 0..1 : 64-row half
  const int nw   = wid & 3;                   // 0..3 : 16-col slice

  const __bf16* x_t    = x    + (size_t)t   * x_tstride;
  const __bf16* wih_d  = wih  + (size_t)dir * G4 * K1;
  const __bf16* whh_d  = whh  + (size_t)dir * G4 * H_;
  const __bf16* h_in_d = h_in + (size_t)dir * B_ * H_;

  // accumulators: [gate][m-subtile], init with per-column bias (b[0]+b[1])
  v8f acc[4][4];
  {
    const float* bp = bias + dir * G4;
#pragma unroll
    for (int g = 0; g < 4; ++g) {
      float bv = bp[g * H_ + nh_base + nw * 16 + (lane & 15)];
      v8f a;
#pragma unroll
      for (int i = 0; i < 8; ++i) a[i] = bv;
#pragma unroll
      for (int mt = 0; mt < 4; ++mt) acc[g][mt] = a;
    }
  }

  const int Ktot = K1 + H_;                   // fused input-proj + recurrent K
  for (int k0 = 0; k0 < Ktot; k0 += 32) {
    const __bf16* asrc; int ald;
    const __bf16* bsrc; int bld;
    int kk;
    if (k0 < K1) { asrc = x_t;    ald = x_ld; bsrc = wih_d; bld = K1; kk = k0; }
    else         { asrc = h_in_d; ald = H_;   bsrc = whh_d; bld = H_; kk = k0 - K1; }

    __syncthreads();
    // A tile: 128 rows x 32 (uint4 = 8 bf16), 2 chunks/thread
#pragma unroll
    for (int i = 0; i < 2; ++i) {
      int c = tid + i * 256;
      int row = c >> 2, seg = c & 3;
      uint4 d = *reinterpret_cast<const uint4*>(asrc + (size_t)(bm + row) * ald + kk + seg * 8);
      *reinterpret_cast<uint4*>(&lds_a[row * LDT + seg * 8]) = d;
    }
    // B tile: 4 gates x 64 weight-rows x 32, 4 chunks/thread
#pragma unroll
    for (int i = 0; i < 4; ++i) {
      int c = tid + i * 256;
      int rl = c >> 2, seg = c & 3;
      int g = rl >> 6, rr = rl & 63;
      int n = g * H_ + nh_base + rr;
      uint4 d = *reinterpret_cast<const uint4*>(bsrc + (size_t)n * bld + kk + seg * 8);
      *reinterpret_cast<uint4*>(&lds_b[rl * LDT + seg * 8]) = d;
    }
    __syncthreads();

    v16bf afr[4];
#pragma unroll
    for (int mt = 0; mt < 4; ++mt)
      afr[mt] = lds_frag_a(&lds_a[(mw * 64 + mt * 16) * LDT]);
#pragma unroll
    for (int g = 0; g < 4; ++g) {
      v16bf bfr = lds_frag_b(&lds_b[(g * 64 + nw * 16) * LDT]);
#pragma unroll
      for (int mt = 0; mt < 4; ++mt)
        acc[g][mt] = __builtin_amdgcn_wmma_f32_16x16x32_bf16(
            false, afr[mt], false, bfr, (short)0, acc[g][mt], false, false);
    }
  }

  // epilogue: all 4 gates of (m, col) live in this wave -> c/h update in-register
  const int col = nh_base + nw * 16 + (lane & 15);
  const int mh  = (lane >> 4) << 3;           // C/D layout: lanes 16-31 hold M+8
  __bf16* h_out_d = h_out + (size_t)dir * B_ * H_;
  float*  c_d     = c_buf + (size_t)dir * B_ * H_;
#pragma unroll
  for (int mt = 0; mt < 4; ++mt) {
#pragma unroll
    for (int v = 0; v < 8; ++v) {
      int m = bm + mw * 64 + mt * 16 + mh + v;
      size_t ci = (size_t)m * H_ + col;
      float xi = acc[0][mt][v];
      float xf = acc[1][mt][v];
      float xg = acc[2][mt][v];
      float xo = acc[3][mt][v];
      float cn = sigf(xf) * c_d[ci] + sigf(xi) * tanhf(xg);
      float hh = sigf(xo) * tanhf(cn);
      c_d[ci] = cn;
      h_out_d[ci] = (__bf16)hh;
      if (out_bf)
        out_bf[((size_t)m * W_ + t) * (2 * H_) + dir * H_ + col] = (__bf16)hh;
      if (out_f && (qlen[m] - 1 == t))
        out_f[(size_t)m * (2 * H_) + dir * H_ + col] = hh;
    }
  }
}

// ---------------- host launcher ----------------
extern "C" void kernel_launch(void* const* d_in, const int* in_sizes, int n_in,
                              void* d_out, int out_size, void* d_ws, size_t ws_size,
                              hipStream_t stream) {
  const int*   qv   = (const int*)d_in[0];
  const int*   qlen = (const int*)d_in[1];
  const float* lut  = (const float*)d_in[2];
  const float* wih0 = (const float*)d_in[3];
  const float* whh0 = (const float*)d_in[4];
  const float* b0   = (const float*)d_in[5];
  const float* wih1 = (const float*)d_in[6];
  const float* whh1 = (const float*)d_in[7];
  const float* b1   = (const float*)d_in[8];
  float* out = (float*)d_out;

  char* ws = (char*)d_ws;
  size_t off = 0;
  auto alloc = [&](size_t bytes) -> void* {
    void* p = ws + off;
    off = (off + bytes + 255) & ~(size_t)255;
    return p;
  };
  __bf16* emb    = (__bf16*)alloc((size_t)B_ * W_ * E_ * 2);        // 20 MB
  __bf16* out0   = (__bf16*)alloc((size_t)B_ * W_ * 2 * H_ * 2);    // 42 MB
  __bf16* wih0b  = (__bf16*)alloc((size_t)2 * G4 * E_ * 2);
  __bf16* whh0b  = (__bf16*)alloc((size_t)2 * G4 * H_ * 2);
  __bf16* wih1b  = (__bf16*)alloc((size_t)2 * G4 * 2 * H_ * 2);
  __bf16* whh1b  = (__bf16*)alloc((size_t)2 * G4 * H_ * 2);
  float*  bias0  = (float*)alloc((size_t)2 * G4 * 4);
  float*  bias1  = (float*)alloc((size_t)2 * G4 * 4);
  __bf16* h_ping = (__bf16*)alloc((size_t)2 * B_ * H_ * 2);
  __bf16* h_pong = (__bf16*)alloc((size_t)2 * B_ * H_ * 2);
  float*  cbuf   = (float*)alloc((size_t)2 * B_ * H_ * 4);

  // weight / bias prep (weights then live bf16 in L2: 23 MB << 192 MB)
  int n;
  n = 2 * G4 * E_;      f32_to_bf16_kernel<<<n / 256, 256, 0, stream>>>(wih0, wih0b, n);
  n = 2 * G4 * H_;      f32_to_bf16_kernel<<<n / 256, 256, 0, stream>>>(whh0, whh0b, n);
  n = 2 * G4 * 2 * H_;  f32_to_bf16_kernel<<<n / 256, 256, 0, stream>>>(wih1, wih1b, n);
  n = 2 * G4 * H_;      f32_to_bf16_kernel<<<n / 256, 256, 0, stream>>>(whh1, whh1b, n);
  bias_combine_kernel<<<(2 * G4 + 255) / 256, 256, 0, stream>>>(b0, bias0);
  bias_combine_kernel<<<(2 * G4 + 255) / 256, 256, 0, stream>>>(b1, bias1);
  embed_kernel<<<B_ * W_, 256, 0, stream>>>(qv, lut, emb);

  const dim3 grid(8, 4, 2);   // nh-tiles, m-tiles, directions
  const int hz = (2 * B_ * H_ * 2) / 4;   // h buffer in u32
  const int cz = 2 * B_ * H_;             // c buffer in u32

  // ---- layer 0 (K1 = E = 512) ----
  zero_u32_kernel<<<hz / 256, 256, 0, stream>>>((unsigned int*)h_ping, hz);
  zero_u32_kernel<<<cz / 256, 256, 0, stream>>>((unsigned int*)cbuf, cz);
  for (int s = 0; s < W_; ++s) {
    const __bf16* hin = (s & 1) ? h_pong : h_ping;
    __bf16*       hou = (s & 1) ? h_ping : h_pong;
    lstm_step_kernel<<<grid, 256, 0, stream>>>(
        emb, E_, W_ * E_, E_,
        wih0b, whh0b, bias0, hin, hou, cbuf,
        out0, (float*)nullptr, (const int*)nullptr, s);
  }

  // ---- layer 1 (K1 = 2H = 1024), writes final rows straight to d_out ----
  zero_u32_kernel<<<hz / 256, 256, 0, stream>>>((unsigned int*)h_ping, hz);
  zero_u32_kernel<<<cz / 256, 256, 0, stream>>>((unsigned int*)cbuf, cz);
  for (int s = 0; s < W_; ++s) {
    const __bf16* hin = (s & 1) ? h_pong : h_ping;
    __bf16*       hou = (s & 1) ? h_ping : h_pong;
    lstm_step_kernel<<<grid, 256, 0, stream>>>(
        out0, 2 * H_, W_ * 2 * H_, 2 * H_,
        wih1b, whh1b, bias1, hin, hou, cbuf,
        (__bf16*)nullptr, out, qlen, s);
  }
}
